// TransolverCrossAttention_ShareKV_56916906606978
// MI455X (gfx1250) — compile-verified, matrix operand order checked by reading, not run on hardware
//
#include <hip/hip_runtime.h>

// ---------------------------------------------------------------------------
// TransolverCrossAttention_ShareKV for MI455X (gfx1250)
// All heavy GEMMs run on v_wmma_f32_16x16x32_bf16 (f32 accumulate).
// B=4, N=16384, D=512, MODES=256, H=8, DH=64, OUT=512
// ---------------------------------------------------------------------------

#define Bsz   4
#define Nn    16384
#define Dd    512
#define MODES 256
#define Hh    8
#define DH    64
#define OUT   512

#if defined(__has_builtin)
#if __has_builtin(__builtin_amdgcn_global_load_async_to_lds_b64) && \
    __has_builtin(__builtin_amdgcn_s_wait_asynccnt)
#define USE_ASYNC_LDS 1
#endif
#endif

typedef __attribute__((ext_vector_type(16))) __bf16       v16bf;
typedef __attribute__((ext_vector_type(8)))  float        v8f;
typedef __attribute__((ext_vector_type(8)))  unsigned int v8u;
typedef __attribute__((ext_vector_type(2)))  int          v2i;   // async b64 payload

__device__ __forceinline__ unsigned short f2bf(float f) {
    unsigned int u = __builtin_bit_cast(unsigned int, f);
    unsigned int r = u + 0x7FFFu + ((u >> 16) & 1u);   // round-to-nearest-even
    return (unsigned short)(r >> 16);
}

// Load one WMMA 16x32 bf16 fragment (A or B operand). Source is row-major
// (line, k) with `stride` u16 elements between lines, k contiguous.
// ISA layout: lanes 0-15 line=lane kb=0; lanes 16-31 line=lane-16 kb=8;
// regs 0-3: K=kb+2r..; regs 4-7: K=16+kb+2r.  (compiler fuses to 2x b128)
__device__ __forceinline__ v16bf load_frag(const unsigned short* base, int stride,
                                           int line, int kb) {
    const unsigned short* p = base + (size_t)line * stride + kb;
    v8u u;
#pragma unroll
    for (int r = 0; r < 4; ++r) {
        u[r]     = *(const unsigned int*)(p + 2 * r);
        u[r + 4] = *(const unsigned int*)(p + 16 + 2 * r);
    }
    return __builtin_bit_cast(v16bf, u);
}

__device__ __forceinline__ v8f wmma_bf16(v16bf a, v16bf b, v8f c) {
    return __builtin_amdgcn_wmma_f32_16x16x32_bf16(
        false, a, false, b, (short)0, c, false, false);
}

// ---------------------------------------------------------------------------
__global__ void k_cvt_bf16(const float* __restrict__ in, unsigned short* __restrict__ out, int n) {
    int i = blockIdx.x * 256 + threadIdx.x;
    if (i < n) out[i] = f2bf(in[i]);
}

__global__ void k_zero_f32(float* __restrict__ p, int n) {
    int i = blockIdx.x * 256 + threadIdx.x;
    if (i < n) p[i] = 0.0f;
}

// ---------------------------------------------------------------------------
// LayerNorm rows of width 512, f32 in -> bf16 out. Optional per-row pre-scale
// 1/(rsum[row]+1e-6) (folds the reference's inv_norm; LN applied after, same
// order as reference). One block (256 thr) per row.
__global__ void k_ln_bf16(const float* __restrict__ x, const float* __restrict__ gamma,
                          const float* __restrict__ beta, unsigned short* __restrict__ out,
                          const float* __restrict__ rsum) {
    __shared__ float red[2][8];
    size_t row = blockIdx.x;
    const float* xr = x + row * Dd;
    int t = threadIdx.x;
    float sc = rsum ? (1.f / (rsum[row] + 1e-6f)) : 1.f;
    float a0 = xr[t] * sc, a1 = xr[t + 256] * sc;
    float s = a0 + a1, s2 = a0 * a0 + a1 * a1;
#pragma unroll
    for (int off = 16; off; off >>= 1) {
        s  += __shfl_xor(s, off, 32);
        s2 += __shfl_xor(s2, off, 32);
    }
    int w = t >> 5;
    if ((t & 31) == 0) { red[0][w] = s; red[1][w] = s2; }
    __syncthreads();
    if (w == 0) {
        float ss  = (t < 8) ? red[0][t] : 0.f;
        float ss2 = (t < 8) ? red[1][t] : 0.f;
#pragma unroll
        for (int off = 4; off; off >>= 1) {
            ss  += __shfl_xor(ss, off, 32);
            ss2 += __shfl_xor(ss2, off, 32);
        }
        if (t == 0) { red[0][0] = ss; red[1][0] = ss2; }
    }
    __syncthreads();
    float mean = red[0][0] * (1.f / Dd);
    float var  = red[1][0] * (1.f / Dd) - mean * mean;
    float rinv = rsqrtf(var + 1e-5f);
    out[row * Dd + t]       = f2bf((a0 - mean) * rinv * gamma[t] + beta[t]);
    out[row * Dd + t + 256] = f2bf((a1 - mean) * rinv * gamma[t + 256] + beta[t + 256]);
}

// ---------------------------------------------------------------------------
// Generic batched GEMM: C[b] = A[b] (RxK bf16 row-major) * W[b]^T
// (W stored CxK row-major -> K contiguous on both operands) + bias.
// Block = 256 thr (8 waves). Wave w: row-tile rt=w&1, col-group cg=w>>1
// -> block computes 32 rows x 256 cols; each wave does 4 wmma per K step
// reusing one A fragment. Optional split-K (nSplit>1): blockIdx.z encodes
// (batch, kslice); results accumulated with f32 atomics (bias must be null).
// A strip is staged in LDS via double-buffered async loads when available.
__global__ void k_gemm_nt(const unsigned short* __restrict__ A,
                          const unsigned short* __restrict__ W,
                          const float* __restrict__ bias,
                          float* __restrict__ outF,
                          unsigned short* __restrict__ outB,
                          int K, int C,
                          long aBS, long wBS, long oBS,
                          int nSplit, int kChunk) {
    int b    = blockIdx.z / nSplit;
    int sk   = blockIdx.z % nSplit;
    int kBeg = sk * kChunk;
    int kEnd = kBeg + kChunk;
    int row0 = blockIdx.y * 32;
    int wave = threadIdx.x >> 5;
    int lane = threadIdx.x & 31;
    int rt   = wave & 1;
    int cg   = wave >> 1;
    int col0 = blockIdx.x * 256 + cg * 64;
    const unsigned short* Ab = A + (size_t)b * aBS + (size_t)row0 * K;
    const unsigned short* Ww = W + (size_t)b * wBS + (size_t)col0 * K;
    int line = lane & 15, kb = (lane >> 4) * 8;
    v8f acc[4] = {};

#ifdef USE_ASYNC_LDS
    __shared__ unsigned short ldsA[2][32 * 32];
    int arow = threadIdx.x >> 3;            // 32 rows
    int ach  = (threadIdx.x & 7) * 4;       // 8 chunks of 8B (4 u16) per row
    {
        const unsigned short* g = Ab + (size_t)arow * K + kBeg + ach;
        __builtin_amdgcn_global_load_async_to_lds_b64(
            (v2i*)g, (v2i*)&ldsA[0][arow * 32 + ach], 0, 0);
    }
    __builtin_amdgcn_s_wait_asynccnt(0);
    __syncthreads();
#endif

    for (int k0 = kBeg; k0 < kEnd; k0 += 32) {
#ifdef USE_ASYNC_LDS
        int buf = ((k0 - kBeg) >> 5) & 1;
        if (k0 + 32 < kEnd) {
            const unsigned short* g = Ab + (size_t)arow * K + k0 + 32 + ach;
            __builtin_amdgcn_global_load_async_to_lds_b64(
                (v2i*)g, (v2i*)&ldsA[buf ^ 1][arow * 32 + ach], 0, 0);
        }
        v16bf a = load_frag(&ldsA[buf][rt * 16 * 32], 32, line, kb);
#else
        v16bf a = load_frag(Ab + (size_t)rt * 16 * K + k0, K, line, kb);
#endif
        __builtin_prefetch((const void*)(Ww + (size_t)line * K + k0 + 128), 0, 1);
#pragma unroll
        for (int j = 0; j < 4; ++j) {
            v16bf w = load_frag(Ww + (size_t)j * 16 * K + k0, K, line, kb);
            acc[j] = wmma_bf16(a, w, acc[j]);
        }
#ifdef USE_ASYNC_LDS
        __builtin_amdgcn_s_wait_asynccnt(0);
        __syncthreads();
#endif
    }

#pragma unroll
    for (int j = 0; j < 4; ++j) {
        int col  = col0 + j * 16 + line;
        float bv = bias ? bias[col] : 0.f;
#pragma unroll
        for (int r = 0; r < 8; ++r) {
            int row   = row0 + rt * 16 + ((lane < 16) ? r : (8 + r));
            float v   = acc[j][r] + bv;
            size_t ix = (size_t)b * oBS + (size_t)row * C + col;
            if (nSplit > 1)      atomicAdd(outF + ix, v);
            else {
                if (outF) outF[ix] = v;
                if (outB) outB[ix] = f2bf(v);
            }
        }
    }
}

// ---------------------------------------------------------------------------
// Softmax over MODES=256 logits per row + mass weighting + column sums.
// Block = 256 thr = 8 waves; one wave per row. grid = (N/8, B).
__global__ void k_softmax_mass(const float* __restrict__ logits,
                               const float* __restrict__ mass,
                               unsigned short* __restrict__ tf,    // may be null
                               unsigned short* __restrict__ tfm,
                               float* __restrict__ colsum) {
    __shared__ float cacc[MODES];
    int t = threadIdx.x;
    cacc[t] = 0.f;
    __syncthreads();
    int wave = t >> 5, lane = t & 31;
    int b = blockIdx.y;
    size_t rowg = (size_t)blockIdx.x * 8 + wave;
    const float* lr = logits + ((size_t)b * Nn + rowg) * MODES;
    float v[8];
    float mx = -1e30f;
#pragma unroll
    for (int j = 0; j < 8; ++j) { v[j] = lr[lane + 32 * j]; mx = fmaxf(mx, v[j]); }
#pragma unroll
    for (int off = 16; off; off >>= 1) mx = fmaxf(mx, __shfl_xor(mx, off, 32));
    float sum = 0.f;
#pragma unroll
    for (int j = 0; j < 8; ++j) { v[j] = __expf(v[j] - mx); sum += v[j]; }
#pragma unroll
    for (int off = 16; off; off >>= 1) sum += __shfl_xor(sum, off, 32);
    float inv = 1.f / sum;
    float ms  = mass[(size_t)b * Nn + rowg];
    size_t ob = ((size_t)b * Nn + rowg) * MODES;
#pragma unroll
    for (int j = 0; j < 8; ++j) {
        int c    = lane + 32 * j;
        float p  = v[j] * inv;
        float pm = p * ms;
        if (tf) tf[ob + c] = f2bf(p);
        tfm[ob + c] = f2bf(pm);
        atomicAdd(&cacc[c], pm);
    }
    __syncthreads();
    atomicAdd(&colsum[b * MODES + t], cacc[t]);
}

// ---------------------------------------------------------------------------
// Tiled transposes (32x32 tiles via LDS). in (R x C) -> out (C x R) per batch.
__global__ void k_transpose_f2b(const float* __restrict__ in, unsigned short* __restrict__ out,
                                int R, int C) {
    __shared__ unsigned short tile[32][33];
    int b = blockIdx.z;
    const float* ib = in + (size_t)b * R * C;
    unsigned short* ob = out + (size_t)b * R * C;
    int c0 = blockIdx.x * 32, r0 = blockIdx.y * 32;
    int tx = threadIdx.x, ty = threadIdx.y;
#pragma unroll
    for (int i = 0; i < 4; ++i)
        tile[ty + 8 * i][tx] = f2bf(ib[(size_t)(r0 + ty + 8 * i) * C + c0 + tx]);
    __syncthreads();
#pragma unroll
    for (int i = 0; i < 4; ++i)
        ob[(size_t)(c0 + ty + 8 * i) * R + r0 + tx] = tile[tx][ty + 8 * i];
}

__global__ void k_transpose_b2b(const unsigned short* __restrict__ in,
                                unsigned short* __restrict__ out, int R, int C) {
    __shared__ unsigned short tile[32][33];
    int b = blockIdx.z;
    const unsigned short* ib = in + (size_t)b * R * C;
    unsigned short* ob = out + (size_t)b * R * C;
    int c0 = blockIdx.x * 32, r0 = blockIdx.y * 32;
    int tx = threadIdx.x, ty = threadIdx.y;
#pragma unroll
    for (int i = 0; i < 4; ++i)
        tile[ty + 8 * i][tx] = ib[(size_t)(r0 + ty + 8 * i) * C + c0 + tx];
    __syncthreads();
#pragma unroll
    for (int i = 0; i < 4; ++i)
        ob[(size_t)(c0 + ty + 8 * i) * R + r0 + tx] = tile[tx][ty + 8 * i];
}

// ---------------------------------------------------------------------------
// Attention over M=256 tokens. Block = 128 thr (4 waves), one 16-query strip.
// grid = (M/16, H, B). S = Q Kh^T / 8 ; softmax ; O = attn * Vh.
__global__ void k_attn(const unsigned short* __restrict__ qh,
                       const unsigned short* __restrict__ kh,
                       const unsigned short* __restrict__ vh,
                       unsigned short* __restrict__ o) {
    __shared__ float          S[16 * MODES];
    __shared__ unsigned short At[16 * MODES];
    __shared__ unsigned short Vt[4][16 * 32];
    __shared__ float          red[16][8];
    int b = blockIdx.z, h = blockIdx.y, m0 = blockIdx.x * 16;
    int wave = threadIdx.x >> 5, lane = threadIdx.x & 31;
    int line = lane & 15, kb = (lane >> 4) * 8;
    const unsigned short* qb = qh + ((size_t)b * MODES + m0) * (Hh * DH) + h * DH;
    const unsigned short* kbse = kh + (size_t)b * MODES * (Hh * DH) + h * DH;

    for (int ct = wave; ct < 16; ct += 4) {
        v8f acc = {};
        const unsigned short* kt = kbse + (size_t)ct * 16 * (Hh * DH);
#pragma unroll
        for (int k0 = 0; k0 < DH; k0 += 32) {
            v16bf a  = load_frag(qb + k0, Hh * DH, line, kb);
            v16bf bm = load_frag(kt + k0, Hh * DH, line, kb);
            acc = wmma_bf16(a, bm, acc);
        }
#pragma unroll
        for (int r = 0; r < 8; ++r) {
            int row = (lane < 16) ? r : (8 + r);
            S[row * MODES + ct * 16 + line] = acc[r] * 0.125f;
        }
    }
    __syncthreads();

    {
        int row = threadIdx.x & 15, seg = threadIdx.x >> 4;
        float* Sr = S + row * MODES + seg * 32;
        float mx = -1e30f;
        for (int j = 0; j < 32; ++j) mx = fmaxf(mx, Sr[j]);
        red[row][seg] = mx;
        __syncthreads();
        float m = red[row][0];
#pragma unroll
        for (int s2 = 1; s2 < 8; ++s2) m = fmaxf(m, red[row][s2]);
        __syncthreads();
        float sum = 0.f;
        for (int j = 0; j < 32; ++j) { float ev = __expf(Sr[j] - m); Sr[j] = ev; sum += ev; }
        red[row][seg] = sum;
        __syncthreads();
        float tot = 0.f;
#pragma unroll
        for (int s2 = 0; s2 < 8; ++s2) tot += red[row][s2];
        float inv = 1.f / tot;
        unsigned short* Ar = At + row * MODES + seg * 32;
        for (int j = 0; j < 32; ++j) Ar[j] = f2bf(Sr[j] * inv);
    }
    __syncthreads();

    v8f acc = {};
    int d0 = wave * 16;
    for (int k0 = 0; k0 < MODES; k0 += 32) {
#pragma unroll
        for (int i = 0; i < 16; ++i) {   // per-wave V^T staging [dd][n]
            int e = lane + 32 * i;
            int n = e >> 4, dd = e & 15;
            Vt[wave][dd * 32 + n] =
                vh[((size_t)b * MODES + k0 + n) * (Hh * DH) + h * DH + d0 + dd];
        }
        v16bf a  = load_frag(At + k0, MODES, line, kb);
        v16bf bm = load_frag(Vt[wave], 32, line, kb);
        acc = wmma_bf16(a, bm, acc);
    }
#pragma unroll
    for (int r = 0; r < 8; ++r) {
        int row = m0 + ((lane < 16) ? r : (8 + r));
        o[((size_t)b * MODES + row) * (Hh * DH) + h * DH + d0 + line] = f2bf(acc[r]);
    }
}

// ---------------------------------------------------------------------------
extern "C" void kernel_launch(void* const* d_in, const int* in_sizes, int n_in,
                              void* d_out, int out_size, void* d_ws, size_t ws_size,
                              hipStream_t stream) {
    (void)in_sizes; (void)n_in; (void)out_size; (void)ws_size;
    const float* qx      = (const float*)d_in[0];
    const float* kvx     = (const float*)d_in[1];
    const float* mass_q  = (const float*)d_in[3];
    const float* mass_kv = (const float*)d_in[4];
    const float* ln_q_g  = (const float*)d_in[5];
    const float* ln_q_b  = (const float*)d_in[6];
    const float* Wlin_q  = (const float*)d_in[7];
    const float* blin_q  = (const float*)d_in[8];
    const float* ln_kv_g = (const float*)d_in[9];
    const float* ln_kv_b = (const float*)d_in[10];
    const float* Wlin_kv = (const float*)d_in[11];
    const float* blin_kv = (const float*)d_in[12];
    const float* nq_g    = (const float*)d_in[13];
    const float* nq_b    = (const float*)d_in[14];
    const float* nkv_g   = (const float*)d_in[15];
    const float* nkv_b   = (const float*)d_in[16];
    const float* Wq      = (const float*)d_in[17];
    const float* bq      = (const float*)d_in[18];
    const float* Wk      = (const float*)d_in[19];
    const float* bk      = (const float*)d_in[20];
    const float* Wv      = (const float*)d_in[21];
    const float* bv      = (const float*)d_in[22];
    const float* Wo      = (const float*)d_in[23];
    const float* bo      = (const float*)d_in[24];
    float* dout = (float*)d_out;

    char* ws = (char*)d_ws;
    size_t off = 0;
    auto alloc = [&](size_t bytes) { void* p = ws + off; off += (bytes + 255) & ~(size_t)255; return p; };
    unsigned short* wWlq  = (unsigned short*)alloc((size_t)MODES * Dd * 2);
    unsigned short* wWlkv = (unsigned short*)alloc((size_t)MODES * Dd * 2);
    unsigned short* wWq   = (unsigned short*)alloc((size_t)Dd * Dd * 2);
    unsigned short* wWk   = (unsigned short*)alloc((size_t)Dd * Dd * 2);
    unsigned short* wWv   = (unsigned short*)alloc((size_t)Dd * Dd * 2);
    unsigned short* wWo   = (unsigned short*)alloc((size_t)Dd * Dd * 2);
    // big buffer reused: LN output (nx) for logits GEMM, then xT (D x N bf16)
    unsigned short* nx     = (unsigned short*)alloc((size_t)Bsz * Nn * Dd * 2);     // 67 MB
    float*          logits = (float*)alloc((size_t)Bsz * Nn * MODES * 4);           // 67 MB (reused q->kv)
    unsigned short* tfm_q  = (unsigned short*)alloc((size_t)Bsz * Nn * MODES * 2);  // 33.5 MB
    unsigned short* tf_kv  = (unsigned short*)alloc((size_t)Bsz * Nn * MODES * 2);
    unsigned short* tfm_kv = (unsigned short*)alloc((size_t)Bsz * Nn * MODES * 2);
    unsigned short* tfmT   = (unsigned short*)alloc((size_t)Bsz * Nn * MODES * 2);  // reused q->kv
    float* colsum_q  = (float*)alloc((size_t)Bsz * MODES * 4);
    float* colsum_kv = (float*)alloc((size_t)Bsz * MODES * 4);
    float* proj_q    = (float*)alloc((size_t)Bsz * MODES * Dd * 4);
    float* proj_kv   = (float*)alloc((size_t)Bsz * MODES * Dd * 4);
    unsigned short* qln  = (unsigned short*)alloc((size_t)Bsz * MODES * Dd * 2);
    unsigned short* kvln = (unsigned short*)alloc((size_t)Bsz * MODES * Dd * 2);
    unsigned short* qhb  = (unsigned short*)alloc((size_t)Bsz * MODES * Dd * 2);
    unsigned short* khb  = (unsigned short*)alloc((size_t)Bsz * MODES * Dd * 2);
    unsigned short* vhb  = (unsigned short*)alloc((size_t)Bsz * MODES * Dd * 2);
    unsigned short* obuf = (unsigned short*)alloc((size_t)Bsz * MODES * Dd * 2);
    float*          yprj = (float*)alloc((size_t)Bsz * MODES * OUT * 4);
    unsigned short* yT   = (unsigned short*)alloc((size_t)Bsz * OUT * MODES * 2);

    dim3 blkT(32, 8);

    // ---- weight conversion & zeroing ---------------------------------------
    k_cvt_bf16<<<(MODES * Dd + 255) / 256, 256, 0, stream>>>(Wlin_q,  wWlq,  MODES * Dd);
    k_cvt_bf16<<<(MODES * Dd + 255) / 256, 256, 0, stream>>>(Wlin_kv, wWlkv, MODES * Dd);
    k_cvt_bf16<<<(Dd * Dd + 255) / 256, 256, 0, stream>>>(Wq, wWq, Dd * Dd);
    k_cvt_bf16<<<(Dd * Dd + 255) / 256, 256, 0, stream>>>(Wk, wWk, Dd * Dd);
    k_cvt_bf16<<<(Dd * Dd + 255) / 256, 256, 0, stream>>>(Wv, wWv, Dd * Dd);
    k_cvt_bf16<<<(Dd * Dd + 255) / 256, 256, 0, stream>>>(Wo, wWo, Dd * Dd);
    k_zero_f32<<<(Bsz * MODES + 255) / 256, 256, 0, stream>>>(colsum_q,  Bsz * MODES);
    k_zero_f32<<<(Bsz * MODES + 255) / 256, 256, 0, stream>>>(colsum_kv, Bsz * MODES);
    k_zero_f32<<<(Bsz * MODES * Dd + 255) / 256, 256, 0, stream>>>(proj_q,  Bsz * MODES * Dd);
    k_zero_f32<<<(Bsz * MODES * Dd + 255) / 256, 256, 0, stream>>>(proj_kv, Bsz * MODES * Dd);

    // ---- subgen: Q side ----------------------------------------------------
    k_ln_bf16<<<Bsz * Nn, 256, 0, stream>>>(qx, ln_q_g, ln_q_b, nx, nullptr);
    k_gemm_nt<<<dim3(MODES / 256, Nn / 32, Bsz), 256, 0, stream>>>(
        nx, wWlq, blin_q, logits, nullptr, Dd, MODES,
        (long)Nn * Dd, 0L, (long)Nn * MODES, 1, Dd);
    k_softmax_mass<<<dim3(Nn / 8, Bsz), 256, 0, stream>>>(
        logits, mass_q, nullptr, tfm_q, colsum_q);
    // proj_q = tfm_q^T @ qx  : pre-transpose both operands, then plain GEMM
    k_transpose_b2b<<<dim3(MODES / 32, Nn / 32, Bsz), blkT, 0, stream>>>(tfm_q, tfmT, Nn, MODES);
    k_transpose_f2b<<<dim3(Dd / 32, Nn / 32, Bsz), blkT, 0, stream>>>(qx, nx, Nn, Dd); // nx = qx^T bf16
    k_gemm_nt<<<dim3(Dd / 256, MODES / 32, Bsz * 8), 256, 0, stream>>>(
        tfmT, nx, nullptr, proj_q, nullptr, Nn, Dd,
        (long)MODES * Nn, (long)Dd * Nn, (long)MODES * Dd, 8, Nn / 8);

    // ---- subgen: KV side (reuses nx / logits / tfmT) -----------------------
    k_ln_bf16<<<Bsz * Nn, 256, 0, stream>>>(kvx, ln_kv_g, ln_kv_b, nx, nullptr);
    k_gemm_nt<<<dim3(MODES / 256, Nn / 32, Bsz), 256, 0, stream>>>(
        nx, wWlkv, blin_kv, logits, nullptr, Dd, MODES,
        (long)Nn * Dd, 0L, (long)Nn * MODES, 1, Dd);
    k_softmax_mass<<<dim3(Nn / 8, Bsz), 256, 0, stream>>>(
        logits, mass_kv, tf_kv, tfm_kv, colsum_kv);
    k_transpose_b2b<<<dim3(MODES / 32, Nn / 32, Bsz), blkT, 0, stream>>>(tfm_kv, tfmT, Nn, MODES);
    k_transpose_f2b<<<dim3(Dd / 32, Nn / 32, Bsz), blkT, 0, stream>>>(kvx, nx, Nn, Dd);
    k_gemm_nt<<<dim3(Dd / 256, MODES / 32, Bsz * 8), 256, 0, stream>>>(
        tfmT, nx, nullptr, proj_kv, nullptr, Nn, Dd,
        (long)MODES * Nn, (long)Dd * Nn, (long)MODES * Dd, 8, Nn / 8);

    // ---- post-projection LayerNorms (inv_norm folded as pre-scale) ---------
    k_ln_bf16<<<Bsz * MODES, 256, 0, stream>>>(proj_q,  nq_g,  nq_b,  qln,  colsum_q);
    k_ln_bf16<<<Bsz * MODES, 256, 0, stream>>>(proj_kv, nkv_g, nkv_b, kvln, colsum_kv);

    // ---- MHA ---------------------------------------------------------------
    k_gemm_nt<<<dim3(Dd / 256, MODES / 32, Bsz), 256, 0, stream>>>(
        qln, wWq, bq, nullptr, qhb, Dd, Dd, (long)MODES * Dd, 0L, (long)MODES * Dd, 1, Dd);
    k_gemm_nt<<<dim3(Dd / 256, MODES / 32, Bsz), 256, 0, stream>>>(
        kvln, wWk, bk, nullptr, khb, Dd, Dd, (long)MODES * Dd, 0L, (long)MODES * Dd, 1, Dd);
    k_gemm_nt<<<dim3(Dd / 256, MODES / 32, Bsz), 256, 0, stream>>>(
        kvln, wWv, bv, nullptr, vhb, Dd, Dd, (long)MODES * Dd, 0L, (long)MODES * Dd, 1, Dd);
    k_attn<<<dim3(MODES / 16, Hh, Bsz), 128, 0, stream>>>(qhb, khb, vhb, obuf);
    k_gemm_nt<<<dim3(OUT / 256, MODES / 32, Bsz), 256, 0, stream>>>(
        obuf, wWo, bo, yprj, nullptr, Dd, OUT, (long)MODES * Dd, 0L, (long)MODES * OUT, 1, Dd);

    // ---- reconstruction ----------------------------------------------------
    k_transpose_f2b<<<dim3(OUT / 32, MODES / 32, Bsz), blkT, 0, stream>>>(yprj, yT, MODES, OUT);
    k_gemm_nt<<<dim3(OUT / 256, Nn / 32, Bsz), 256, 0, stream>>>(
        tf_kv, yT, nullptr, dout, nullptr, MODES, OUT,
        (long)Nn * MODES, (long)OUT * MODES, (long)Nn * OUT, 1, MODES);
}
